// _Head_56006373540300
// MI455X (gfx1250) — matrix operand order repaired and checked by reading, not compiled, not run on hardware
//
#include <hip/hip_runtime.h>
#include <hip/hip_bf16.h>

// ---------------------------------------------------------------------------
// Causal single-head attention, CDNA5 (gfx1250, wave32, WMMA 16x16x32 f16,
// TDM tensor_load_to_lds double-buffered staging for K/V tiles).
//   out = softmax(mask(q k^T * hs^-0.5)) v,  q=xWq, k=xWk, v=xWv
// B=8, T=2048, C=1024, HS=64.
// ---------------------------------------------------------------------------

typedef __attribute__((ext_vector_type(16))) _Float16 v16h;
typedef __attribute__((ext_vector_type(8)))  _Float16 v8h;
typedef __attribute__((ext_vector_type(8)))  float    v8f;
typedef __attribute__((ext_vector_type(4)))  float    v4f;
typedef __attribute__((ext_vector_type(4)))  unsigned int v4u;
typedef __attribute__((ext_vector_type(8)))  int      v8i;
typedef __attribute__((ext_vector_type(4)))  int      v4i;

#define B_  8
#define T_  2048
#define C_  1024
#define HS_ 64

static __device__ __forceinline__ v16h cat16(v8h a, v8h b) {
  return __builtin_shufflevector(a, b, 0,1,2,3,4,5,6,7,8,9,10,11,12,13,14,15);
}

static __device__ __forceinline__ v8f wmma_f16(v16h a, v16h b, v8f c) {
  // D = A(16x32 f16) * B(32x16 f16) + C(16x16 f32)
  return __builtin_amdgcn_wmma_f32_16x16x32_f16(
      /*neg_a=*/false, a, /*neg_b=*/false, b,
      /*c_mod=*/(short)0, c, /*reuse_a=*/false, /*reuse_b=*/false);
}

// ---------------------------------------------------------------------------
// TDM: async 2-D tile load Global -> LDS (f16 elements).
// D# per CDNA5 ISA ch.8: group0 = {count=1 | lds_addr | global_addr | type=2},
// group1 = {data_size=1(2B), tensor_dim0/1, tile_dim0/1, dim0_stride}.
// Tracked with TENSORcnt. 6-arg builtin form (clang-23 toolchain).
// ---------------------------------------------------------------------------
static __device__ __forceinline__ void tdm_load_2d_f16(
    unsigned lds_addr, const _Float16* gptr,
    unsigned tensor_d0, unsigned tensor_d1, unsigned stride0,
    unsigned tile_d0, unsigned tile_d1) {
  unsigned long long ga = (unsigned long long)(const void*)gptr;
  v4u g0;
  g0[0] = 1u;                                           // count=1, user descriptor
  g0[1] = lds_addr;                                     // LDS byte address
  g0[2] = (unsigned)(ga & 0xffffffffu);                 // global addr [31:0]
  g0[3] = (unsigned)((ga >> 32) & 0x01ffffffu)          // global addr [56:32]
        | (2u << 30);                                   // type = 2 ("image")
  v8i g1;
  g1[0] = (int)(1u << 16);                              // data_size = 1 -> 2 bytes
  g1[1] = (int)((tensor_d0 & 0xffffu) << 16);           // tensor_dim0 lo16
  g1[2] = (int)(((tensor_d0 >> 16) & 0xffffu) | ((tensor_d1 & 0xffffu) << 16));
  g1[3] = (int)(((tensor_d1 >> 16) & 0xffffu) | ((tile_d0 & 0xffffu) << 16));
  g1[4] = (int)(tile_d1 & 0xffffu);                     // tile_dim1 (tile_dim2=0)
  g1[5] = (int)stride0;                                 // tensor_dim0_stride lo32
  g1[6] = 0;                                            // stride hi / dim1_stride
  g1[7] = 0;
  v4i g2 = {0, 0, 0, 0};                                // dims 2/3 unused (2-D tile)
  v4i g3 = {0, 0, 0, 0};
  v8i g4 = {0, 0, 0, 0, 0, 0, 0, 0};                    // extra group (6-arg form)
  __builtin_amdgcn_tensor_load_to_lds(g0, g1, g2, g3, g4, 0);
}

static __device__ __forceinline__ unsigned lds_byte_addr(const void* p) {
  // LDS aperture: flat addr[31:0] maps to the LDS byte offset.
  return (unsigned)(size_t)p;
}

// ---------------------------------------------------------------------------
// Kernel 1: repack W (fp32 [C,HS]) into f16 WMMA B-fragment panels.
// Panel layout: [mat:3][kblock:C/32][ntile:4][lane:32] x v16h.
// B-fragment (16-bit, 32x16): lane = {half=lane>>4, n=lane&15},
// element e<8 = B[half*8 + e][n], e>=8 = B[16 + half*8 + e-8][n].
// ---------------------------------------------------------------------------
__global__ void repack_w(const float* __restrict__ Wq,
                         const float* __restrict__ Wk,
                         const float* __restrict__ Wv,
                         _Float16* __restrict__ Wpan) {
  int tid  = blockIdx.x * 256 + threadIdx.x;   // 3*32*4*32 = 12288 threads
  int lane = tid & 31;
  int nt   = (tid >> 5) & 3;
  int kb   = (tid >> 7) & 31;
  int w    = tid >> 12;                        // 0=q, 1=k, 2=v
  const float* W = (w == 0) ? Wq : ((w == 1) ? Wk : Wv);
  int half = lane >> 4, n = lane & 15;
  v16h frag;
#pragma unroll
  for (int e = 0; e < 16; ++e)
    frag[e] = (_Float16)W[(kb * 32 + (e < 8 ? half * 8 + e : 16 + half * 8 + (e - 8))) * HS_ + nt * 16 + n];
  ((v16h*)Wpan)[((w * 32 + kb) * 4 + nt) * 32 + lane] = frag;
}

// ---------------------------------------------------------------------------
// Kernel 2: fused q/k/v projection. One wave per 16-token tile; x is read
// exactly once and converted to f16 A-fragments on the fly.
// Writes: qh [B*T,HS] f16 (pre-scaled by HS^-0.5), kh [B*T,HS] f16,
//         vT [B][HS][T] f16 (transposed -> contiguous B-fragments later).
// ---------------------------------------------------------------------------
__global__ void qkv_proj(const float* __restrict__ x,
                         const _Float16* __restrict__ Wpan,
                         _Float16* __restrict__ qh,
                         _Float16* __restrict__ kh,
                         _Float16* __restrict__ vT) {
  int wave = threadIdx.x >> 5, lane = threadIdx.x & 31;
  int half = lane >> 4, mr = lane & 15;
  int m0   = (blockIdx.x * 4 + wave) * 16;          // global token-row
  const float* xrow = x + (size_t)(m0 + mr) * C_;

  v8f acc[3][4];
  v8f z = {0.f, 0.f, 0.f, 0.f, 0.f, 0.f, 0.f, 0.f};
#pragma unroll
  for (int w = 0; w < 3; ++w)
#pragma unroll
    for (int nt = 0; nt < 4; ++nt) acc[w][nt] = z;

  const v16h* wp = (const v16h*)Wpan;
  for (int k0 = 0; k0 < C_; k0 += 32) {
    // A-fragment (16-bit 16x32): elems 0..7 = K = half*8..+7,
    //                            elems 8..15 = K = 16+half*8..+7
    v4f f0 = *(const v4f*)(xrow + k0 + half * 8);
    v4f f1 = *(const v4f*)(xrow + k0 + half * 8 + 4);
    v4f f2 = *(const v4f*)(xrow + k0 + 16 + half * 8);
    v4f f3 = *(const v4f*)(xrow + k0 + 16 + half * 8 + 4);
    v16h a;
#pragma unroll
    for (int e = 0; e < 4; ++e) {
      a[e]      = (_Float16)f0[e];
      a[4 + e]  = (_Float16)f1[e];
      a[8 + e]  = (_Float16)f2[e];
      a[12 + e] = (_Float16)f3[e];
    }
    int kb = k0 >> 5;
#pragma unroll
    for (int w = 0; w < 3; ++w) {
#pragma unroll
      for (int nt = 0; nt < 4; ++nt) {
        v16h bf = wp[((w * 32 + kb) * 4 + nt) * 32 + lane];
        acc[w][nt] = wmma_f16(a, bf, acc[w][nt]);
      }
    }
  }

  // Epilogue. C/D layout: VGPR r -> row M = r + 8*half, col N = lane&15.
  int b  = m0 / T_;
  int t0 = m0 - b * T_;
#pragma unroll
  for (int nt = 0; nt < 4; ++nt) {
#pragma unroll
    for (int r = 0; r < 8; ++r) {
      size_t row = (size_t)(m0 + r + 8 * half);
      qh[row * HS_ + nt * 16 + mr] = (_Float16)(acc[0][nt][r] * 0.125f); // fold hs^-0.5
      kh[row * HS_ + nt * 16 + mr] = (_Float16)(acc[1][nt][r]);
    }
    v8h pv;
#pragma unroll
    for (int r = 0; r < 8; ++r) pv[r] = (_Float16)acc[2][nt][r];
    *(v8h*)(vT + ((size_t)b * HS_ + nt * 16 + mr) * T_ + t0 + 8 * half) = pv;
  }
}

// ---------------------------------------------------------------------------
// Kernel 3: flash attention. One wave owns a 16-row query tile; causal key
// loop in blocks of 32. K (32x64) and V^T (64x32) tiles are DMA'd into LDS
// by the Tensor Data Mover, double-buffered, tracked with TENSORcnt.
// Online softmax stats in f32; P goes through a per-wave LDS patch to
// convert C-layout -> A-layout for the P*V WMMAs.
// ---------------------------------------------------------------------------
__global__ void flash_attn(const _Float16* __restrict__ qh,
                           const _Float16* __restrict__ kh,
                           const _Float16* __restrict__ vT,
                           float* __restrict__ out) {
  __shared__ __align__(32) _Float16 kst[4][2][32 * 64];  // [wave][buf] K tile
  __shared__ __align__(32) _Float16 vst[4][2][64 * 32];  // [wave][buf] V^T tile
  __shared__ __align__(32) _Float16 plds[4][16 * 32];    // P C->A transpose patch
  int wave = threadIdx.x >> 5, lane = threadIdx.x & 31;
  int half = lane >> 4, mr = lane & 15;
  int b  = blockIdx.x >> 5;                    // 32 chunks of 64 rows per batch
  int m0 = (blockIdx.x & 31) * 64 + wave * 16; // token row within batch
  const _Float16* qb = qh + (size_t)b * T_ * HS_;
  const _Float16* kb = kh + (size_t)b * T_ * HS_;
  const _Float16* vb = vT + (size_t)b * HS_ * T_;
  _Float16* pl = &plds[wave][0];

  // q A-fragments for K-blocks h=0..31 and h=32..63 (row fixed per lane)
  const _Float16* qrow = qb + (size_t)(m0 + mr) * HS_;
  v16h aq0 = cat16(*(const v8h*)(qrow + half * 16),
                   *(const v8h*)(qrow + half * 16 + 8));
  v16h aq1 = cat16(*(const v8h*)(qrow + 32 + half * 16),
                   *(const v8h*)(qrow + 32 + half * 16 + 8));

  v8f z = {0.f, 0.f, 0.f, 0.f, 0.f, 0.f, 0.f, 0.f};
  v8f o[4];
#pragma unroll
  for (int nt = 0; nt < 4; ++nt) o[nt] = z;
  float mi[8], li[8];
#pragma unroll
  for (int r = 0; r < 8; ++r) { mi[r] = -1e30f; li[r] = 0.f; }

  int nblk = (m0 + 16) >> 5;                   // ceil((m0+16)/32): m0+16 mult of 16
  if ((m0 + 16) & 31) nblk++;

  // Prologue: DMA block 0 into buffer 0 (K: 32 rows x 64 h; V^T: 64 h x 32 t)
  tdm_load_2d_f16(lds_byte_addr(&kst[wave][0][0]), kb, HS_, T_, HS_, HS_, 32);
  tdm_load_2d_f16(lds_byte_addr(&vst[wave][0][0]), vb, T_, HS_, T_, 32, HS_);

  for (int bi = 0; bi < nblk; ++bi) {
    int j0  = bi * 32;
    int cur = bi & 1;
    if (bi + 1 < nblk) {                       // prefetch next block's tiles
      tdm_load_2d_f16(lds_byte_addr(&kst[wave][1 - cur][0]),
                      kb + (size_t)(j0 + 32) * HS_, HS_, T_, HS_, HS_, 32);
      tdm_load_2d_f16(lds_byte_addr(&vst[wave][1 - cur][0]),
                      vb + (j0 + 32), T_, HS_, T_, 32, HS_);
      __builtin_amdgcn_s_wait_tensorcnt(2);    // current buffer's 2 DMAs done
    } else {
      __builtin_amdgcn_s_wait_tensorcnt(0);
    }
    asm volatile("" ::: "memory");             // keep ds reads below the wait

    const _Float16* kt = &kst[wave][cur][0];
    const _Float16* vt = &vst[wave][cur][0];

    // S = q k^T : two 16x16 tiles covering keys j0..j0+31 (B-frags from LDS)
    v8f s[2];
#pragma unroll
    for (int jn = 0; jn < 2; ++jn) {
      const _Float16* krow = kt + (jn * 16 + mr) * HS_;
      v16h bk0 = *(const v16h*)(krow + half * 16);       // h = 0..31
      v16h bk1 = *(const v16h*)(krow + 32 + half * 16);  // h = 32..63
      v8f c = wmma_f16(aq0, bk0, z);
      s[jn]  = wmma_f16(aq1, bk1, c);
    }
    // causal mask + online softmax (C-layout: col = mr, row = m0+r+8*half)
#pragma unroll
    for (int r = 0; r < 8; ++r) {
      int row  = m0 + r + 8 * half;
      float s0 = (j0 + mr      <= row) ? s[0][r] : -1e30f;
      float s1 = (j0 + 16 + mr <= row) ? s[1][r] : -1e30f;
      float rm = fmaxf(s0, s1);
#pragma unroll
      for (int off = 1; off < 16; off <<= 1)
        rm = fmaxf(rm, __shfl_xor(rm, off, 32));         // stays in 16-lane half
      float mnew = fmaxf(mi[r], rm);
      float p0 = __expf(s0 - mnew);
      float p1 = __expf(s1 - mnew);
      float lad = p0 + p1;
#pragma unroll
      for (int off = 1; off < 16; off <<= 1)
        lad += __shfl_xor(lad, off, 32);
      float alpha = __expf(mi[r] - mnew);
      li[r] = li[r] * alpha + lad;
      mi[r] = mnew;
#pragma unroll
      for (int nt = 0; nt < 4; ++nt) o[nt][r] *= alpha;
      pl[(r + 8 * half) * 32 + mr]      = (_Float16)p0;  // C-layout -> LDS
      pl[(r + 8 * half) * 32 + 16 + mr] = (_Float16)p1;
    }
    // P back as A-fragment (16x32), then O += P * V
    v16h aP = cat16(*(const v8h*)(pl + mr * 32 + half * 16),
                    *(const v8h*)(pl + mr * 32 + half * 16 + 8));
#pragma unroll
    for (int nt = 0; nt < 4; ++nt) {
      v16h bv = *(const v16h*)(vt + (nt * 16 + mr) * 32 + half * 16);
      o[nt] = wmma_f16(aP, bv, o[nt]);
    }
  }

  // out = O / l  (fp32, [B,T,HS])
#pragma unroll
  for (int nt = 0; nt < 4; ++nt) {
#pragma unroll
    for (int r = 0; r < 8; ++r) {
      out[((size_t)b * T_ + m0 + r + 8 * half) * HS_ + nt * 16 + mr] =
          o[nt][r] / li[r];
    }
  }
}

// ---------------------------------------------------------------------------
extern "C" void kernel_launch(void* const* d_in, const int* in_sizes, int n_in,
                              void* d_out, int out_size, void* d_ws, size_t ws_size,
                              hipStream_t stream) {
  (void)in_sizes; (void)n_in; (void)out_size; (void)ws_size;
  const float* x  = (const float*)d_in[0];
  // d_in[1] = attention_mask: pure causal tril -> computed arithmetically, never read.
  const float* Wk = (const float*)d_in[2];
  const float* Wq = (const float*)d_in[3];
  const float* Wv = (const float*)d_in[4];
  float* out = (float*)d_out;

  char* ws = (char*)d_ws;
  _Float16* qh = (_Float16*)(ws);                      // 2 MB
  _Float16* kh = (_Float16*)(ws + (2u << 20));         // 2 MB
  _Float16* vT = (_Float16*)(ws + (4u << 20));         // 2 MB
  _Float16* Wp = (_Float16*)(ws + (6u << 20));         // 384 KB

  repack_w<<<48, 256, 0, stream>>>(Wq, Wk, Wv, Wp);
  qkv_proj<<<(B_ * T_) / 64, 128, 0, stream>>>(x, Wp, qh, kh, vT);
  flash_attn<<<B_ * (T_ / 64), 128, 0, stream>>>(qh, kh, vT, out);
}
